// RNN_8237747274192
// MI455X (gfx1250) — matrix operand order repaired and checked by reading, not compiled, not run on hardware
//
#include <hip/hip_runtime.h>

typedef __attribute__((ext_vector_type(16))) __bf16 v16bf;
typedef __attribute__((ext_vector_type(8)))  float  v8f;

#define B_  256
#define T_  2048
#define I_  64
#define H_  256

#define BM   16            // batch rows per workgroup
#define NT   2             // 16-wide N tiles per wave (wave covers 32 cols)
#define KS   10            // k-steps per timestep: 2 (input, K=64) + 8 (recurrent, K=256)
#define HPAD (H_ + 8)      // pad to break LDS bank conflicts
#define IPAD (I_ + 8)

__global__ __launch_bounds__(256)
void rnn_persistent_wmma(const float* __restrict__ x,
                         const int*   __restrict__ lengths,
                         const float* __restrict__ W_ih,
                         const float* __restrict__ W_hh,
                         const float* __restrict__ b_ih,
                         const float* __restrict__ b_hh,
                         const float* __restrict__ W_fc,
                         const float* __restrict__ b_fc,
                         float*       __restrict__ out)
{
    __shared__ __bf16 hs[BM][HPAD];     // h_{t-1} (bf16, shared A operand)
    __shared__ __bf16 xs[BM][IPAD];     // x_t staged (bf16)
    __shared__ float  hlast[BM][H_];    // h at t = len-1 (f32)
    __shared__ float  bias_s[H_];       // b_ih + b_hh
    __shared__ int    len_s[BM];
    __shared__ float  red[BM][16];

    const int tid  = threadIdx.x;
    const int wave = tid >> 5;
    const int lane = tid & 31;
    const int half = lane >> 4;         // wmma lane half
    const int l16  = lane & 15;
    const int m0   = blockIdx.x * BM;

    // ---------------- one-time setup ----------------
    if (tid < H_) bias_s[tid] = b_ih[tid] + b_hh[tid];
    if (tid < BM) len_s[tid]  = lengths[m0 + tid];
    for (int idx = tid; idx < BM * HPAD; idx += 256)
        ((__bf16*)hs)[idx] = (__bf16)0.0f;

    // Persistent B-operand fragments in VGPRs.
    // B frag k order used here (contiguous, same as A): lanes 0-15 hold K=0..15 (elems 0..15),
    // lanes 16-31 hold K=16..31, for column n = lane&15.
    // B[k][n] = W[n_global][k]  (computing h @ W^T), so each lane streams
    // 16 contiguous elements of its own W row.
    v16bf whh[NT][8];   // K = 256 -> 8 k-steps of 32
    v16bf wih[NT][2];   // K = 64  -> 2 k-steps of 32
    #pragma unroll
    for (int nt = 0; nt < NT; ++nt) {
        const int row = wave * 32 + nt * 16 + l16;
        #pragma unroll
        for (int f = 0; f < 8; ++f) {
            const int kb = f * 32 + half * 16;
            const float* wr = W_hh + (size_t)row * H_ + kb;
            #pragma unroll
            for (int e = 0; e < 16; ++e) whh[nt][f][e] = (__bf16)wr[e];
        }
        #pragma unroll
        for (int f = 0; f < 2; ++f) {
            const int kb = f * 32 + half * 16;
            const float* wr = W_ih + (size_t)row * I_ + kb;
            #pragma unroll
            for (int e = 0; e < 16; ++e) wih[nt][f][e] = (__bf16)wr[e];
        }
    }

    // per-lane bias, hoisted out of the recurrence (folded in at tanh)
    float bv[NT];
    #pragma unroll
    for (int nt = 0; nt < NT; ++nt) bv[nt] = bias_s[wave * 32 + nt * 16 + l16];

    // stage x_0
    {
        const int m = tid >> 4, i0 = (tid & 15) * 4;
        const float* xp = x + (size_t)(m0 + m) * T_ * I_ + i0;
        #pragma unroll
        for (int j = 0; j < 4; ++j) xs[m][i0 + j] = (__bf16)xp[j];
    }
    __syncthreads();

    // ---------------- sequential recurrence ----------------
    #pragma unroll 1
    for (int t = 0; t < T_; ++t) {
        // A-fragment fetch (16-bit 16x32 layout): row m = l16;
        // elems 0..15 = K kb..kb+15 (contiguous);
        // half 1 shifts kb by 16. k-steps 0..1 read xs (K=64), 2..9 read hs (K=256).
        auto loadA = [&](int s) -> v16bf {
            v16bf a;
            if (s < 2) {
                const int kb = s * 32 + half * 16;
                #pragma unroll
                for (int e = 0; e < 8; ++e) {
                    a[e]     = xs[l16][kb + e];
                    a[8 + e] = xs[l16][kb + 8 + e];
                }
            } else {
                const int kb = (s - 2) * 32 + half * 16;
                #pragma unroll
                for (int e = 0; e < 8; ++e) {
                    a[e]     = hs[l16][kb + e];
                    a[8 + e] = hs[l16][kb + 8 + e];
                }
            }
            return a;
        };

        v8f acc[NT] = {};   // zero C -> first WMMA takes inline-0 SRC2

        // software-pipelined k-loop: 3-slot rotating A buffer, loads issued
        // 2 k-steps ahead of consumption so each ds_load pair hides under
        // two WMMA pairs (waits become dscnt<=4 instead of dscnt==0).
        v16bf abuf[3];
        abuf[0] = loadA(0);
        abuf[1] = loadA(1);
        #pragma unroll
        for (int s = 0; s < KS; ++s) {
            if (s + 2 < KS) abuf[(s + 2) % 3] = loadA(s + 2);
            #pragma unroll
            for (int nt = 0; nt < NT; ++nt) {
                const v16bf bf = (s < 2) ? wih[nt][s] : whh[nt][s - 2];
                acc[nt] = __builtin_amdgcn_wmma_f32_16x16x32_bf16(
                    false, abuf[s % 3], false, bf, (short)0, acc[nt], false, false);
            }
        }

        // prefetch next timestep's x while finishing this step
        if (t + 1 < T_) {
            const int m = tid >> 4, i0 = (tid & 15) * 4;
            __builtin_prefetch(x + (size_t)(m0 + m) * T_ * I_ + (size_t)(t + 1) * I_ + i0, 0, 1);
        }

        // bias + nonlinearity (in place, f32)
        #pragma unroll
        for (int nt = 0; nt < NT; ++nt)
            #pragma unroll
            for (int e = 0; e < 8; ++e) acc[nt][e] = tanhf(acc[nt][e] + bv[nt]);

        __syncthreads();   // everyone done reading hs/xs of step t

        // write h_t back to LDS (+ capture last valid h in f32)
        #pragma unroll
        for (int nt = 0; nt < NT; ++nt) {
            const int n = wave * 32 + nt * 16 + l16;
            #pragma unroll
            for (int e = 0; e < 8; ++e) {
                const int m = half * 8 + e;
                const float hv = acc[nt][e];
                hs[m][n] = (__bf16)hv;
                if (len_s[m] == t + 1) hlast[m][n] = hv;
            }
        }

        // stage x_{t+1}
        if (t + 1 < T_) {
            const int m = tid >> 4, i0 = (tid & 15) * 4;
            const float* xp = x + (size_t)(m0 + m) * T_ * I_ + (size_t)(t + 1) * I_ + i0;
            #pragma unroll
            for (int j = 0; j < 4; ++j) xs[m][i0 + j] = (__bf16)xp[j];
        }
        __syncthreads();
    }

    // ---------------- final FC: out[b] = hlast[b,:]·W_fc + b_fc ----------------
    {
        const int m = tid >> 4, seg = tid & 15;
        float s = 0.0f;
        #pragma unroll
        for (int j = 0; j < 16; ++j)
            s += hlast[m][seg * 16 + j] * W_fc[seg * 16 + j];
        red[m][seg] = s;
    }
    __syncthreads();
    if (tid < BM) {
        float s = b_fc[0];
        #pragma unroll
        for (int j = 0; j < 16; ++j) s += red[tid][j];
        out[m0 + tid] = s;
    }
}

extern "C" void kernel_launch(void* const* d_in, const int* in_sizes, int n_in,
                              void* d_out, int out_size, void* d_ws, size_t ws_size,
                              hipStream_t stream) {
    (void)in_sizes; (void)n_in; (void)d_ws; (void)ws_size; (void)out_size;
    const float* x       = (const float*)d_in[0];
    const int*   lengths = (const int*)  d_in[1];
    const float* W_ih    = (const float*)d_in[2];
    const float* W_hh    = (const float*)d_in[3];
    const float* b_ih    = (const float*)d_in[4];
    const float* b_hh    = (const float*)d_in[5];
    const float* W_fc    = (const float*)d_in[6];
    const float* b_fc    = (const float*)d_in[7];
    float*       out     = (float*)d_out;

    rnn_persistent_wmma<<<dim3(B_ / BM), dim3(256), 0, stream>>>(
        x, lengths, W_ih, W_hh, b_ih, b_hh, W_fc, b_fc, out);
}
